// MultiHeadAttention_88673894793396
// MI455X (gfx1250) — matrix operand order repaired and checked by reading, not compiled, not run on hardware
//
#include <hip/hip_runtime.h>

#define BATCH 4
#define SEQ   2048
#define DIM   1024
#define HEADS 16
#define HDIM  64
#define MROWS (BATCH * SEQ) /* 8192 */
#define BK    64            /* GEMM K-tile */

typedef __attribute__((ext_vector_type(16))) __bf16 v16bf;
typedef __attribute__((ext_vector_type(8)))  __bf16 v8bf;
typedef __attribute__((ext_vector_type(8)))  float  v8f;

union FragU { v16bf v; v8bf h[2]; };

__device__ __forceinline__ unsigned short f2bf(float f) {
  union { float f; unsigned u; } x; x.f = f;
  unsigned r = x.u + 0x7FFFu + ((x.u >> 16) & 1u); // round-to-nearest-even
  return (unsigned short)(r >> 16);
}

__device__ __forceinline__ v16bf load_frag2(const unsigned short* p0,
                                            const unsigned short* p1) {
  FragU f;
  f.h[0] = *(const v8bf*)p0;
  f.h[1] = *(const v8bf*)p1;
  return f.v;
}

// ----- CDNA5 async global->LDS copy (16B per lane) via inline asm -----
// GLOBAL_LOAD_ASYNC_TO_LDS_B128: VDST = LDS byte address VGPR, VADDR = 64-bit
// global address (GV mode). Tracked by ASYNCcnt; waited with s_wait_asynccnt.
#if defined(__gfx1250__)
#define HAS_ASYNC_LDS 1
#else
#define HAS_ASYNC_LDS 0
#endif

__device__ __forceinline__ void cp_async16(unsigned short* lds_dst,
                                           const unsigned short* gsrc) {
#if HAS_ASYNC_LDS
  typedef __attribute__((address_space(3))) unsigned short lds_us;
  unsigned loff = (unsigned)(unsigned long long)(lds_us*)lds_dst;
  asm volatile("global_load_async_to_lds_b128 %0, %1, off"
               :
               : "v"(loff), "v"(gsrc)
               : "memory");
#else
  *(uint4*)lds_dst = *(const uint4*)gsrc;
#endif
}

__device__ __forceinline__ void async_wait() {
#if HAS_ASYNC_LDS
  asm volatile("s_wait_asynccnt 0x0" ::: "memory");
#endif
}

// ---------------- elementwise conversion kernels ----------------

__global__ void cvt_bf16_kernel(const float* __restrict__ in,
                                unsigned short* __restrict__ out, int n) {
  int i = blockIdx.x * blockDim.x + threadIdx.x;
  if (i < n) out[i] = f2bf(in[i]);
}

// W is (K,N) row-major; emit W^T as (N,K) bf16 so GEMM B-tiles are K-contiguous.
__global__ void cvt_wT_kernel(const float* __restrict__ in,
                              unsigned short* __restrict__ out) {
  int i = blockIdx.x * blockDim.x + threadIdx.x;   // over DIM*DIM
  int k = i >> 10, n = i & 1023;
  out[n * DIM + k] = f2bf(in[i]);
}

// V[B,S,D] -> Vt[B,H,HDIM,S] so the PxV B-fragments are contiguous.
__global__ void vT_kernel(const unsigned short* __restrict__ V,
                          unsigned short* __restrict__ Vt) {
  int i = blockIdx.x * blockDim.x + threadIdx.x;   // over MROWS*DIM
  int d = i & 1023;
  int s = (i >> 10) & 2047;
  int b = i >> 21;
  int hh = d >> 6, hd = d & 63;
  Vt[(((size_t)b * HEADS + hh) * HDIM + hd) * SEQ + s] = V[i];
}

// ---------------- 256x128 double-buffered WMMA GEMM ----------------
// out = A(bf16)[MROWS,DIM] x W + bias, W given as Bt = W^T [DIM(out),DIM(in)] bf16.
// 256 threads = 8 waves in 4(m) x 2(n); each wave: 4x4 tiles of 16x16 (64x64).
// Double-buffered LDS (96KB) with async global->LDS fills overlapped with WMMAs.

template <bool OUTF32>
__global__ __launch_bounds__(256) void gemm256(const unsigned short* __restrict__ A,
                                               const unsigned short* __restrict__ Bt,
                                               const float* __restrict__ bias,
                                               void* __restrict__ outp) {
  __shared__ alignas(16) unsigned short As[2][256 * BK];
  __shared__ alignas(16) unsigned short Bs[2][128 * BK];
  const int tid = threadIdx.x, lane = tid & 31, wid = tid >> 5;
  const int hf = lane >> 4, l15 = lane & 15;
  const int wm = wid & 3, wn = wid >> 2;
  const int m0 = blockIdx.y * 256, n0 = blockIdx.x * 128;

  v8f vz = {};
  v8f acc[4][4];
#pragma unroll
  for (int i = 0; i < 4; ++i)
#pragma unroll
    for (int j = 0; j < 4; ++j) acc[i][j] = vz;

  auto load_tiles = [&](int buf, int k0) {
    for (int c = tid; c < 2048; c += 256) {       // 256 rows x 8 chunks of 8 bf16
      int r = c >> 3, seg = c & 7;
      cp_async16(&As[buf][r * BK + seg * 8],
                 A + (size_t)(m0 + r) * DIM + k0 + seg * 8);
    }
    for (int c = tid; c < 1024; c += 256) {       // 128 rows x 8 chunks
      int r = c >> 3, seg = c & 7;
      cp_async16(&Bs[buf][r * BK + seg * 8],
                 Bt + (size_t)(n0 + r) * DIM + k0 + seg * 8);
    }
  };

  load_tiles(0, 0);
  int cur = 0;
  for (int kt = 0; kt < DIM / BK; ++kt) {
    async_wait();
    __syncthreads();                 // all fills for `cur` visible; prev compute done
    if (kt + 1 < DIM / BK) load_tiles(cur ^ 1, (kt + 1) * BK);

    const unsigned short* Ab = As[cur];
    const unsigned short* Bb = Bs[cur];
#pragma unroll
    for (int ks = 0; ks < 2; ++ks) {
      v16bf af[4], bfr[4];
#pragma unroll
      for (int i = 0; i < 4; ++i) {
        int row = wm * 64 + i * 16 + l15;              // A: lane row = l&15
        af[i] = load_frag2(Ab + row * BK + ks * 32 + hf * 8,
                           Ab + row * BK + ks * 32 + 16 + hf * 8);
      }
#pragma unroll
      for (int j = 0; j < 4; ++j) {
        int nn = wn * 64 + j * 16 + l15;               // B: lane col = l&15
        bfr[j] = load_frag2(Bb + nn * BK + ks * 32 + hf * 8,
                            Bb + nn * BK + ks * 32 + 16 + hf * 8);
      }
#pragma unroll
      for (int i = 0; i < 4; ++i)
#pragma unroll
        for (int j = 0; j < 4; ++j)
          acc[i][j] = __builtin_amdgcn_wmma_f32_16x16x32_bf16(
              false, af[i], false, bfr[j], (short)0, acc[i][j], false, false);
    }
    cur ^= 1;
  }

#pragma unroll
  for (int i = 0; i < 4; ++i)
#pragma unroll
    for (int j = 0; j < 4; ++j)
#pragma unroll
      for (int r = 0; r < 8; ++r) {
        int m = m0 + wm * 64 + i * 16 + r + 8 * hf;   // C layout: M = r + 8*half
        int n = n0 + wn * 64 + j * 16 + l15;
        float v = acc[i][j][r] + bias[n];
        if (OUTF32)
          ((float*)outp)[(size_t)m * DIM + n] = v;
        else
          ((unsigned short*)outp)[(size_t)m * DIM + n] = f2bf(v);
      }
}

// ---------------- flash attention: 4 waves, 64 q-rows per block ----------------
// grid = (SEQ/64, HEADS, BATCH). Q,K: [B,S,D] bf16; Vt: [B,H,HDIM,S] bf16.

__global__ __launch_bounds__(128) void attn_kernel(const unsigned short* __restrict__ Q,
                                                   const unsigned short* __restrict__ K,
                                                   const unsigned short* __restrict__ Vt,
                                                   unsigned short* __restrict__ ctx) {
  __shared__ alignas(16) unsigned short Qs[64 * 64];
  __shared__ alignas(16) unsigned short Ks[64 * 64];
  __shared__ alignas(16) unsigned short Vs[64 * 64]; // Vt tile: [d][kv]
  __shared__ alignas(16) unsigned short Ps[64 * 64]; // P re-layout (C->A), per-wave rows
  const int tid = threadIdx.x, lane = tid & 31, wid = tid >> 5;
  const int hf = lane >> 4, l15 = lane & 15;
  const int qt = blockIdx.x, hh = blockIdx.y, bb = blockIdx.z;
  const int s0 = qt * 64;

  const unsigned short* Qg = Q + ((size_t)bb * SEQ + s0) * DIM + hh * HDIM;
  for (int c = tid; c < 512; c += 128) { // 64 rows x 8 chunks of 8 bf16
    int r = c >> 3, seg = c & 7;
    cp_async16(Qs + r * 64 + seg * 8, Qg + (size_t)r * DIM + seg * 8);
  }
  async_wait();
  __syncthreads();

  v16bf aq[2];
  {
    int row = wid * 16 + l15;
    aq[0] = load_frag2(Qs + row * 64 + hf * 8,      Qs + row * 64 + 16 + hf * 8);
    aq[1] = load_frag2(Qs + row * 64 + 32 + hf * 8, Qs + row * 64 + 48 + hf * 8);
  }

  float m_st[8], l_st[8];
  v8f vz = {};
  v8f o_acc[4];
#pragma unroll
  for (int r = 0; r < 8; ++r) { m_st[r] = -1e30f; l_st[r] = 0.0f; }
#pragma unroll
  for (int j = 0; j < 4; ++j) o_acc[j] = vz;

  const unsigned short* Kg = K + (size_t)bb * SEQ * DIM + hh * HDIM;
  const unsigned short* Vg = Vt + ((size_t)bb * HEADS + hh) * HDIM * SEQ;

  for (int kv0 = 0; kv0 < SEQ; kv0 += 64) {
    __syncthreads();   // previous tile's consumers done
    for (int c = tid; c < 512; c += 128) {
      int r = c >> 3, seg = c & 7;
      cp_async16(Ks + r * 64 + seg * 8, Kg + (size_t)(kv0 + r) * DIM + seg * 8);
      cp_async16(Vs + r * 64 + seg * 8, Vg + (size_t)r * SEQ + kv0 + seg * 8);
    }
    async_wait();
    __syncthreads();

    // scores = Q * K^T ; B-frag element (k=d, n=kv) = K[kv][d] -> contiguous in d
    v8f s_acc[4];
#pragma unroll
    for (int j = 0; j < 4; ++j) s_acc[j] = vz;
#pragma unroll
    for (int ks = 0; ks < 2; ++ks) {
#pragma unroll
      for (int j = 0; j < 4; ++j) {
        int kvc = j * 16 + l15;
        v16bf bk = load_frag2(Ks + kvc * 64 + ks * 32 + hf * 8,
                              Ks + kvc * 64 + ks * 32 + 16 + hf * 8);
        s_acc[j] = __builtin_amdgcn_wmma_f32_16x16x32_bf16(
            false, aq[ks], false, bk, (short)0, s_acc[j], false, false);
      }
    }

    // online softmax (scale = 1/sqrt(64) = 0.125); row M = r + 8*hf in this half
#pragma unroll
    for (int j = 0; j < 4; ++j) s_acc[j] = s_acc[j] * 0.125f;
#pragma unroll
    for (int r = 0; r < 8; ++r) {
      float vm = -1e30f;
#pragma unroll
      for (int j = 0; j < 4; ++j) vm = fmaxf(vm, s_acc[j][r]);
      vm = fmaxf(vm, __shfl_xor(vm, 1, 32));
      vm = fmaxf(vm, __shfl_xor(vm, 2, 32));
      vm = fmaxf(vm, __shfl_xor(vm, 4, 32));
      vm = fmaxf(vm, __shfl_xor(vm, 8, 32));
      float m_new = fmaxf(m_st[r], vm);
      float alpha = __expf(m_st[r] - m_new);
      m_st[r] = m_new;
      float rs = 0.0f;
#pragma unroll
      for (int j = 0; j < 4; ++j) {
        float e = __expf(s_acc[j][r] - m_new);
        Ps[(wid * 16 + r + 8 * hf) * 64 + j * 16 + l15] = f2bf(e);
        rs += e;
      }
      rs += __shfl_xor(rs, 1, 32);
      rs += __shfl_xor(rs, 2, 32);
      rs += __shfl_xor(rs, 4, 32);
      rs += __shfl_xor(rs, 8, 32);
      l_st[r] = l_st[r] * alpha + rs;
#pragma unroll
      for (int j = 0; j < 4; ++j) o_acc[j][r] = o_acc[j][r] * alpha;
    }

    // O += P * V ; A from Ps (per-wave rows, same-wave DS dependency only)
#pragma unroll
    for (int ks = 0; ks < 2; ++ks) {
      int prow = wid * 16 + l15;
      v16bf ap = load_frag2(Ps + prow * 64 + ks * 32 + hf * 8,
                            Ps + prow * 64 + ks * 32 + 16 + hf * 8);
#pragma unroll
      for (int j = 0; j < 4; ++j) {
        int dc = j * 16 + l15;
        v16bf bv = load_frag2(Vs + dc * 64 + ks * 32 + hf * 8,
                              Vs + dc * 64 + ks * 32 + 16 + hf * 8);
        o_acc[j] = __builtin_amdgcn_wmma_f32_16x16x32_bf16(
            false, ap, false, bv, (short)0, o_acc[j], false, false);
      }
    }
  }

  unsigned short* Cg = ctx + ((size_t)bb * SEQ + s0) * DIM + hh * HDIM;
#pragma unroll
  for (int r = 0; r < 8; ++r) {
    float inv = 1.0f / l_st[r];
    int srow = wid * 16 + r + 8 * hf;
#pragma unroll
    for (int j = 0; j < 4; ++j)
      Cg[(size_t)srow * DIM + j * 16 + l15] = f2bf(o_acc[j][r] * inv);
  }
}

// ---------------- launch ----------------

extern "C" void kernel_launch(void* const* d_in, const int* in_sizes, int n_in,
                              void* d_out, int out_size, void* d_ws, size_t ws_size,
                              hipStream_t stream) {
  const float* x  = (const float*)d_in[0];
  const float* Wq = (const float*)d_in[1];
  const float* bq = (const float*)d_in[2];
  const float* Wk = (const float*)d_in[3];
  const float* bk = (const float*)d_in[4];
  const float* Wv = (const float*)d_in[5];
  const float* bv = (const float*)d_in[6];
  const float* Wo = (const float*)d_in[7];
  const float* bo = (const float*)d_in[8];

  // Workspace layout (bf16 = ushort), total ~109 MB.
  unsigned short* p = (unsigned short*)d_ws;
  unsigned short* xb  = p; p += (size_t)MROWS * DIM;
  unsigned short* WqT = p; p += (size_t)DIM * DIM;
  unsigned short* WkT = p; p += (size_t)DIM * DIM;
  unsigned short* WvT = p; p += (size_t)DIM * DIM;
  unsigned short* WoT = p; p += (size_t)DIM * DIM;
  unsigned short* Qb  = p; p += (size_t)MROWS * DIM;
  unsigned short* Kb  = p; p += (size_t)MROWS * DIM;
  unsigned short* Vb  = p; p += (size_t)MROWS * DIM;
  unsigned short* Vt  = p; p += (size_t)MROWS * DIM;
  unsigned short* ctx = p; p += (size_t)MROWS * DIM;

  const int nx = MROWS * DIM; // 8388608
  cvt_bf16_kernel<<<nx / 256, 256, 0, stream>>>(x, xb, nx);
  cvt_wT_kernel<<<(DIM * DIM) / 256, 256, 0, stream>>>(Wq, WqT);
  cvt_wT_kernel<<<(DIM * DIM) / 256, 256, 0, stream>>>(Wk, WkT);
  cvt_wT_kernel<<<(DIM * DIM) / 256, 256, 0, stream>>>(Wv, WvT);
  cvt_wT_kernel<<<(DIM * DIM) / 256, 256, 0, stream>>>(Wo, WoT);

  dim3 gg(DIM / 128, MROWS / 256);
  gemm256<false><<<gg, 256, 0, stream>>>(xb, WqT, bq, (void*)Qb);
  gemm256<false><<<gg, 256, 0, stream>>>(xb, WkT, bk, (void*)Kb);
  gemm256<false><<<gg, 256, 0, stream>>>(xb, WvT, bv, (void*)Vb);

  vT_kernel<<<nx / 256, 256, 0, stream>>>(Vb, Vt);

  dim3 ga(SEQ / 64, HEADS, BATCH);
  attn_kernel<<<ga, 128, 0, stream>>>(Qb, Kb, Vt, ctx);

  gemm256<true><<<gg, 256, 0, stream>>>(ctx, WoT, bo, d_out);
}
